// LSTM_9560597201288
// MI455X (gfx1250) — matrix operand order repaired and checked by reading, not compiled, not run on hardware
//
#include <hip/hip_runtime.h>

// ---------------------------------------------------------------------------
// LSTM (I=256, H=1024, O=256, B=64, T=512) for MI455X / gfx1250.
// bf16 WMMA (v_wmma_f32_16x16x32_bf16) with f32 accumulation, gates fused.
// Depth-2 software pipeline pinned with sched_barrier; async global->LDS
// staging via global_load_async_to_lds_b128 + s_wait_asynccnt.
// ---------------------------------------------------------------------------

typedef __attribute__((ext_vector_type(16))) __bf16 v16bf;
typedef __attribute__((ext_vector_type(8)))  float  v8f;

union ABf { v16bf v; uint4 u[2]; };

#define I_DIM 256
#define H_DIM 1024
#define O_DIM 256
#define B_DIM 64
#define T_DIM 512
#define KCAT  1280                     // I + H
#define KT_N  (KCAT / 32)              // 40 k-tiles in the recurrent GEMM
#define LDS_STRIDE 1288                // 1280 + 8 bf16 pad -> row stride 2576B (conflict-free)
#define STEP_SMEM (B_DIM * LDS_STRIDE * 2)  // 164,864 B dynamic LDS (< 320KB/WGP)

#if defined(__gfx1250__) && __has_builtin(__builtin_amdgcn_global_load_async_to_lds_b128)
#define USE_ASYNC_LDS 1
typedef int v4i_vs __attribute__((vector_size(4 * sizeof(int))));
typedef __attribute__((address_space(1))) v4i_vs GlobalV4;
typedef __attribute__((address_space(3))) v4i_vs LdsV4;
#else
#define USE_ASYNC_LDS 0
#endif

#if __has_builtin(__builtin_amdgcn_sched_barrier)
#define SCHED_FENCE() __builtin_amdgcn_sched_barrier(0)
#else
#define SCHED_FENCE()
#endif

__device__ __forceinline__ float sigmoidf_(float v) { return 1.0f / (1.0f + __expf(-v)); }

#define WMMA_BF16(A, Bm, C) \
  __builtin_amdgcn_wmma_f32_16x16x32_bf16(false, (A), false, (Bm), (short)0, (C), false, false)

// A fragment: two 16B chunks 32B apart (CDNA5 16-bit A 16x32 layout, per-lane)
__device__ __forceinline__ void load_a(ABf& a, const __bf16* p) {
  a.u[0] = *(const uint4*)(p);
  a.u[1] = *(const uint4*)(p + 16);
}
// B fragment: two 16B chunks 32B apart (same k order as A)
__device__ __forceinline__ void load_b(ABf& b, const __bf16* p) {
  b.u[0] = *(const uint4*)(p);
  b.u[1] = *(const uint4*)(p + 16);
}

// ---------------------------------------------------------------------------
// Prep: Wcat = bf16([W_ih | W_hh]) row-major [4096][1280]; Wout bf16 [256][1024];
//       bias = b_ih + b_hh; zero h0 and c.
// ---------------------------------------------------------------------------
__global__ void lstm_prep(const float* __restrict__ W_ih, const float* __restrict__ W_hh,
                          const float* __restrict__ b_ih, const float* __restrict__ b_hh,
                          const float* __restrict__ W_out,
                          __bf16* __restrict__ Wcat, __bf16* __restrict__ Wout,
                          float* __restrict__ bias, __bf16* __restrict__ h0,
                          float* __restrict__ c)
{
  const int blk = blockIdx.x;
  const int t   = threadIdx.x;              // 256 threads
  if (blk < 4 * H_DIM) {                    // one Wcat row per block
    const float* sih = W_ih + (size_t)blk * I_DIM;
    const float* shh = W_hh + (size_t)blk * H_DIM;
    __bf16* dst = Wcat + (size_t)blk * KCAT;
    dst[t] = (__bf16)sih[t];                // t covers 0..255 == I_DIM
    for (int k = t; k < H_DIM; k += 256) dst[I_DIM + k] = (__bf16)shh[k];
  } else if (blk < 4 * H_DIM + O_DIM) {     // one Wout row per block
    const int r = blk - 4 * H_DIM;
    for (int k = t; k < H_DIM; k += 256)
      Wout[(size_t)r * H_DIM + k] = (__bf16)W_out[(size_t)r * H_DIM + k];
  } else if (blk == 4 * H_DIM + O_DIM) {
    for (int k = t; k < 4 * H_DIM; k += 256) bias[k] = b_ih[k] + b_hh[k];
  } else if (blk == 4 * H_DIM + O_DIM + 1) {
    for (int k = t; k < B_DIM * H_DIM; k += 256) h0[k] = (__bf16)0.0f;
  } else {
    for (int k = t; k < B_DIM * H_DIM; k += 256) c[k] = 0.0f;
  }
}

// ---------------------------------------------------------------------------
// One recurrent step:  z = [x_s | h_s] @ Wcat^T  (4 gates fused in registers),
// c/h update, plus fused readout of h_s (already staged in LDS):
//   y[:, s-1, :] = h_s @ W_out^T + b_out        (skipped for s==0)
// Grid: 64 workgroups (16 hidden columns each) x 128 threads (4 waves = 4 M-tiles).
// ---------------------------------------------------------------------------
__global__ __launch_bounds__(128) void lstm_step(
    const float*  __restrict__ x,
    const __bf16* __restrict__ Wcat,
    const __bf16* __restrict__ Wout,
    const float*  __restrict__ bias,
    const float*  __restrict__ b_out,
    float*        __restrict__ c,
    const __bf16* __restrict__ h_in,
    __bf16*       __restrict__ h_out,
    float*        __restrict__ y,
    int s)
{
  extern __shared__ __align__(16) char smem_raw[];
  __bf16* lds = (__bf16*)smem_raw;          // Xcat: [64 rows][1288 cols] bf16 (1280 used)

  const int tid  = threadIdx.x;
  const int wave = tid >> 5;
  const int lane = tid & 31;
  const int nblk = blockIdx.x;              // 0..63 -> hidden columns [nblk*16, +16)

  // ---- stage Xcat = [bf16(x[:,s,:]) | h_in] into LDS ----
  for (int g = tid; g < (B_DIM * I_DIM) / 4; g += 128) {
    const int row  = g >> 6;                // 64 groups of 4 floats per row
    const int col4 = (g & 63) << 2;
    const float4 v = *(const float4*)(x + ((size_t)row * T_DIM + s) * I_DIM + col4);
    union { __bf16 b[4]; uint2 u; } p;
    p.b[0] = (__bf16)v.x; p.b[1] = (__bf16)v.y;
    p.b[2] = (__bf16)v.z; p.b[3] = (__bf16)v.w;
    *(uint2*)(lds + row * LDS_STRIDE + col4) = p.u;
  }
#if USE_ASYNC_LDS
  for (int g = tid; g < (B_DIM * H_DIM) / 8; g += 128) {
    const int row  = g >> 7;                // 128 groups of 8 bf16 per row
    const int col8 = (g & 127) << 3;
    __builtin_amdgcn_global_load_async_to_lds_b128(
        (GlobalV4*)(h_in + row * H_DIM + col8),
        (LdsV4*)(lds + row * LDS_STRIDE + I_DIM + col8),
        0, 0);
  }
#if __has_builtin(__builtin_amdgcn_s_wait_asynccnt)
  __builtin_amdgcn_s_wait_asynccnt(0);
#else
  asm volatile("s_wait_asynccnt 0" ::: "memory");
#endif
#else
  for (int g = tid; g < (B_DIM * H_DIM) / 8; g += 128) {
    const int row  = g >> 7;                // 128 groups of 8 bf16 per row
    const int col8 = (g & 127) << 3;
    const uint4 v = *(const uint4*)(h_in + row * H_DIM + col8);
    *(uint4*)(lds + row * LDS_STRIDE + I_DIM + col8) = v;
  }
#endif
  __syncthreads();

  // ---- fragment addressing (per CDNA5 WMMA VGPR layouts) ----
  const int l15  = lane & 15;
  const int lgrp = lane >> 4;
  const int ak0  = lgrp << 3;               // A: 8-elem chunk offset within K-tile (0|8)
  const int bk0  = lgrp << 3;               // B: 8-elem chunk offset within K-tile (0|8)

  const __bf16* lA   = lds + ((wave << 4) + l15) * LDS_STRIDE + ak0;
  const int     brow = (nblk << 4) + l15;
  const __bf16* pBI  = Wcat + (size_t)(0 * H_DIM + brow) * KCAT + bk0;
  const __bf16* pBF  = Wcat + (size_t)(1 * H_DIM + brow) * KCAT + bk0;
  const __bf16* pBG  = Wcat + (size_t)(2 * H_DIM + brow) * KCAT + bk0;
  const __bf16* pBO  = Wcat + (size_t)(3 * H_DIM + brow) * KCAT + bk0;

  v8f accI = {0.f,0.f,0.f,0.f,0.f,0.f,0.f,0.f};
  v8f accF = {0.f,0.f,0.f,0.f,0.f,0.f,0.f,0.f};
  v8f accG = {0.f,0.f,0.f,0.f,0.f,0.f,0.f,0.f};
  v8f accO = {0.f,0.f,0.f,0.f,0.f,0.f,0.f,0.f};

  // ---- depth-2 software pipeline, pinned with sched barriers:
  //      loads(kt+2) | fence | wmma(kt) | fence  => two k-steps always in flight
  ABf a[3], bi[3], bf[3], bg[3], bo[3];
  #pragma unroll
  for (int p = 0; p < 2; ++p) {
    const int ke = p << 5;
    load_a(a[p], lA + ke);
    load_b(bi[p], pBI + ke); load_b(bf[p], pBF + ke);
    load_b(bg[p], pBG + ke); load_b(bo[p], pBO + ke);
  }
  #pragma unroll
  for (int kt = 0; kt < KT_N; ++kt) {
    SCHED_FENCE();
    if (kt + 2 < KT_N) {
      const int nb = (kt + 2) % 3;
      const int ke = (kt + 2) << 5;
      load_a(a[nb], lA + ke);
      load_b(bi[nb], pBI + ke); load_b(bf[nb], pBF + ke);
      load_b(bg[nb], pBG + ke); load_b(bo[nb], pBO + ke);
    }
    SCHED_FENCE();
    const int cb = kt % 3;
    accI = WMMA_BF16(a[cb].v, bi[cb].v, accI);
    accF = WMMA_BF16(a[cb].v, bf[cb].v, accF);
    accG = WMMA_BF16(a[cb].v, bg[cb].v, accG);
    accO = WMMA_BF16(a[cb].v, bo[cb].v, accO);
  }
  SCHED_FENCE();

  // ---- gates + state update, straight from WMMA accumulators ----
  // C/D layout: lane l, acc elem e  ->  n = l&15, m = e + 8*(l>>4) within the tile.
  const int   ng = (nblk << 4) + l15;
  const float bI = bias[ng];
  const float bF = bias[H_DIM + ng];
  const float bG = bias[2 * H_DIM + ng];
  const float bO = bias[3 * H_DIM + ng];
  const int   mb = (wave << 4) + (lgrp << 3);
  #pragma unroll
  for (int e = 0; e < 8; ++e) {
    const int   m  = mb + e;
    const float ig = sigmoidf_(accI[e] + bI);
    const float fg = sigmoidf_(accF[e] + bF);
    const float gg = tanhf(accG[e] + bG);
    const float og = sigmoidf_(accO[e] + bO);
    const int   idx = m * H_DIM + ng;
    const float cn  = fg * c[idx] + ig * gg;
    c[idx] = cn;
    h_out[idx] = (__bf16)(og * tanhf(cn));
  }

  // ---- fused readout of h_s (staged in LDS): y[:, s-1, :] ----
  if (wave == 0 && s > 0) {
    const int mt = nblk >> 4;               // 4 M-tiles x 16 N-tiles = 64 workgroups
    const int nt = nblk & 15;
    v8f acc = {0.f,0.f,0.f,0.f,0.f,0.f,0.f,0.f};
    const __bf16* lA2 = lds + ((mt << 4) + l15) * LDS_STRIDE + I_DIM + ak0;
    const __bf16* pB  = Wout + (size_t)((nt << 4) + l15) * H_DIM + bk0;
    ABf ra[3], rb[3];
    #pragma unroll
    for (int p = 0; p < 2; ++p) {
      const int ke = p << 5;
      load_a(ra[p], lA2 + ke); load_b(rb[p], pB + ke);
    }
    #pragma unroll
    for (int kt = 0; kt < H_DIM / 32; ++kt) {
      SCHED_FENCE();
      if (kt + 2 < H_DIM / 32) {
        const int nb = (kt + 2) % 3;
        const int ke = (kt + 2) << 5;
        load_a(ra[nb], lA2 + ke); load_b(rb[nb], pB + ke);
      }
      SCHED_FENCE();
      const int cb = kt % 3;
      acc = WMMA_BF16(ra[cb].v, rb[cb].v, acc);
    }
    SCHED_FENCE();
    const int   o   = (nt << 4) + l15;
    const float bb  = b_out[o];
    const int   mb2 = (mt << 4) + (lgrp << 3);
    #pragma unroll
    for (int e = 0; e < 8; ++e) {
      const int m = mb2 + e;
      y[((size_t)m * T_DIM + (s - 1)) * O_DIM + o] = acc[e] + bb;
    }
  }
}

// ---------------------------------------------------------------------------
// Final readout for h_T: y[:, 511, :] = h_T @ W_out^T + b_out.
// 64 blocks x 1 wave; one 16x16 output tile each, K=1024.
// ---------------------------------------------------------------------------
__global__ __launch_bounds__(32) void lstm_final(
    const __bf16* __restrict__ h,
    const __bf16* __restrict__ Wout,
    const float*  __restrict__ b_out,
    float*        __restrict__ y)
{
  const int lane = threadIdx.x;
  const int mt   = blockIdx.x >> 4;
  const int nt   = blockIdx.x & 15;
  const int l15  = lane & 15;
  const int lgrp = lane >> 4;
  const int ak0  = lgrp << 3;
  const int bk0  = lgrp << 3;

  const __bf16* pA = h    + (size_t)((mt << 4) + l15) * H_DIM + ak0;
  const __bf16* pB = Wout + (size_t)((nt << 4) + l15) * H_DIM + bk0;

  v8f acc = {0.f,0.f,0.f,0.f,0.f,0.f,0.f,0.f};
  ABf ra[3], rb[3];
  #pragma unroll
  for (int p = 0; p < 2; ++p) {
    const int ke = p << 5;
    load_a(ra[p], pA + ke); load_b(rb[p], pB + ke);
  }
  #pragma unroll
  for (int kt = 0; kt < H_DIM / 32; ++kt) {
    SCHED_FENCE();
    if (kt + 2 < H_DIM / 32) {
      const int nb = (kt + 2) % 3;
      const int ke = (kt + 2) << 5;
      load_a(ra[nb], pA + ke); load_b(rb[nb], pB + ke);
    }
    SCHED_FENCE();
    const int cb = kt % 3;
    acc = WMMA_BF16(ra[cb].v, rb[cb].v, acc);
  }
  SCHED_FENCE();
  const int   o  = (nt << 4) + l15;
  const float bb = b_out[o];
  const int   mb = (mt << 4) + (lgrp << 3);
  #pragma unroll
  for (int e = 0; e < 8; ++e) {
    const int m = mb + e;
    y[((size_t)m * T_DIM + (T_DIM - 1)) * O_DIM + o] = acc[e] + bb;
  }
}

// ---------------------------------------------------------------------------
extern "C" void kernel_launch(void* const* d_in, const int* in_sizes, int n_in,
                              void* d_out, int out_size, void* d_ws, size_t ws_size,
                              hipStream_t stream) {
  (void)in_sizes; (void)n_in; (void)out_size; (void)ws_size;
  const float* x     = (const float*)d_in[0];
  const float* W_ih  = (const float*)d_in[1];
  const float* W_hh  = (const float*)d_in[2];
  const float* b_ih  = (const float*)d_in[3];
  const float* b_hh  = (const float*)d_in[4];
  const float* W_out = (const float*)d_in[5];
  const float* b_out = (const float*)d_in[6];
  float* y = (float*)d_out;

  char* ws = (char*)d_ws;
  size_t off = 0;
  auto take = [&](size_t bytes) -> char* {
    char* p = ws + off;
    off = (off + bytes + 255) & ~(size_t)255;
    return p;
  };
  __bf16* Wcat = (__bf16*)take((size_t)4 * H_DIM * KCAT * 2);   // 10.5 MB (L2-resident)
  __bf16* Wout = (__bf16*)take((size_t)O_DIM * H_DIM * 2);      // 512 KB
  float*  bias = (float*) take((size_t)4 * H_DIM * 4);          // 16 KB
  float*  c    = (float*) take((size_t)B_DIM * H_DIM * 4);      // 256 KB
  __bf16* h0   = (__bf16*)take((size_t)B_DIM * H_DIM * 2);      // 128 KB
  __bf16* h1   = (__bf16*)take((size_t)B_DIM * H_DIM * 2);      // 128 KB

  lstm_prep<<<4 * H_DIM + O_DIM + 3, 256, 0, stream>>>(
      W_ih, W_hh, b_ih, b_hh, W_out, Wcat, Wout, bias, h0, c);

  for (int s = 0; s < T_DIM; ++s) {
    __bf16* hi = (s & 1) ? h1 : h0;
    __bf16* ho = (s & 1) ? h0 : h1;
    lstm_step<<<64, 128, STEP_SMEM, stream>>>(
        x, Wcat, Wout, bias, b_out, c, hi, ho, y, s);
  }
  // after s=511 (odd), h_T lives in h0
  lstm_final<<<64, 32, 0, stream>>>(h0, Wout, b_out, y);
}
